// SupConLoss_52647709114635
// MI455X (gfx1250) — matrix operand order repaired and checked by reading, not compiled, not run on hardware
//
#include <hip/hip_runtime.h>
#include <hip/hip_bf16.h>

typedef __attribute__((ext_vector_type(16))) _Float16 v16h;
typedef __attribute__((ext_vector_type(8)))  float    v8f;

#define BATCH 65536
#define DFEAT 1024
#define NAU   64
#define NCLS  8
#define INV_TEMP (1.0f / 0.07f)

// ---------------------------------------------------------------------------
// Kernel 1: reset scratch (last-valid-index accumulator) to -1.
// ---------------------------------------------------------------------------
__global__ void supcon_init_ws(int* ws) { ws[0] = -1; }

// ---------------------------------------------------------------------------
// Kernel 2: find the LAST index b with labels[b] != 0 (the only sample that
// survives the reference's loss-reset bug). Block-local max first, then one
// global atomicMax per block.
// ---------------------------------------------------------------------------
__global__ void supcon_scan_last(const int* __restrict__ labels, int* ws, int n) {
    __shared__ int smax;
    if (threadIdx.x == 0) smax = -1;
    __syncthreads();
    int i = blockIdx.x * blockDim.x + threadIdx.x;
    if (i < n && labels[i] != 0) atomicMax(&smax, i);
    __syncthreads();
    if (threadIdx.x == 0 && smax >= 0) atomicMax(ws, smax);
}

// ---------------------------------------------------------------------------
// Kernel 3: single workgroup (4 waves). Wave w computes dots for AUs
// [16w, 16w+16) via v_wmma_f32_16x16x32_f16 over K = 1024, then thread 0
// finishes the masked log-sum-exp loss for sample b*.
//
// A-matrix (16x32 f16, ISA-documented layout): lane L, half h ->
//   K = 16*(h/8) + 8*(L/16) + (h%8); mot[b*] broadcast into every M row.
// B-matrix (32x16 f16, same K order as A): N = L%16,
//   K = 16*(h/8) + 8*(L/16) + (h%8).
// D (16x16 f32, documented): row M=0 in c[0], lanes 0..15 (all rows equal
// here because A is row-broadcast).
// ---------------------------------------------------------------------------
__global__ void __launch_bounds__(128)
supcon_last_sample(const int* __restrict__ labels,
                   const float* __restrict__ mot,    // [B, D]
                   const float* __restrict__ au,     // [A, D]
                   const int* __restrict__ ex_au,    // [C, A]
                   const int* __restrict__ ws,
                   float* __restrict__ out)
{
    __shared__ float sdots[NAU];

    const int b = ws[0];
    if (b < 0) {                      // uniform across the workgroup
        if (threadIdx.x == 0) out[0] = 0.0f;
        return;
    }

    const int tid    = threadIdx.x;
    const int wave   = tid >> 5;      // 0..3 -> AU tile base = 16*wave
    const int lane   = tid & 31;
    const int laneLo = lane & 15;     // N column for B/D
    const int laneHi = lane >> 4;     // lane-half select

    const float* __restrict__ mrow = mot + (size_t)b * DFEAT;
    const float* __restrict__ arow = au  + (size_t)(wave * 16 + laneLo) * DFEAT;

    v8f c = {};
    for (int d0 = 0; d0 < DFEAT; d0 += 32) {
        v16h aA, bB;
#pragma unroll
        for (int h = 0; h < 16; ++h) {
            const int kA = ((h >> 3) << 4) + (laneHi << 3) + (h & 7);
            aA[h] = (_Float16)mrow[d0 + kA];
            const int kB = ((h >> 3) << 4) + (laneHi << 3) + (h & 7);
            bB[h] = (_Float16)arow[d0 + kB];
        }
        c = __builtin_amdgcn_wmma_f32_16x16x32_f16(
                /*neg_a=*/false, aA, /*neg_b=*/false, bB,
                /*c_mod=*/(short)0, c, /*reuse_a=*/false, /*reuse_b=*/false);
    }

    // Row M=0 of D: c[0] in lanes 0..15 holds <mot[b*], au[a_base+n]>.
    if (laneHi == 0) sdots[wave * 16 + laneLo] = c[0] * INV_TEMP;
    __syncthreads();

    if (tid == 0) {
        const int lab = labels[b];
        const int* __restrict__ mask = ex_au + lab * NAU;
        float neg_sum = 0.0f, pos_sum = 0.0f, n_pos = 0.0f;
        for (int a = 0; a < NAU; ++a) {
            const float d = sdots[a];
            if (mask[a] != 0) { pos_sum += d; n_pos += 1.0f; }
            else              { neg_sum += expf(d); }
        }
        const float denom      = logf(neg_sum);
        const float per_sample = (pos_sum - n_pos * denom) / n_pos;
        out[0] = per_sample / (float)BATCH;
    }
}

// ---------------------------------------------------------------------------
// Host launcher: init -> scan -> compute, all on `stream` (graph-capture safe).
// Inputs (setup_inputs order): labels[int32], mot_feats[f32], au_feats_all[f32],
// ex_au[int32]. Output: 1 float.
// ---------------------------------------------------------------------------
extern "C" void kernel_launch(void* const* d_in, const int* in_sizes, int n_in,
                              void* d_out, int out_size, void* d_ws, size_t ws_size,
                              hipStream_t stream) {
    const int*   labels = (const int*)  d_in[0];
    const float* mot    = (const float*)d_in[1];
    const float* au     = (const float*)d_in[2];
    const int*   ex_au  = (const int*)  d_in[3];
    float* out = (float*)d_out;
    int*   ws  = (int*)d_ws;

    supcon_init_ws<<<1, 1, 0, stream>>>(ws);
    supcon_scan_last<<<(BATCH + 255) / 256, 256, 0, stream>>>(labels, ws, BATCH);
    supcon_last_sample<<<1, 128, 0, stream>>>(labels, mot, au, ex_au, ws, out);
}
